// SpikeSequencePreprocessor_7284264534654
// MI455X (gfx1250) — matrix operand order repaired and checked by reading, not compiled, not run on hardware
//
#include <hip/hip_runtime.h>

// SNN 1x1-conv mix + leaky integrate-and-fire, fused.
//   inputs : spikes f32 [32,512,2048], weight f32 [512,512], bias f32 [512]
//   output : spk f32 [32,512,2048]
// Roofline: 268 MB mandatory traffic -> ~11.5 us @ 23.3 TB/s; GEMM is 34.4
// GFLOP, trivially hidden by f16 WMMA. Fusion avoids a 256 MiB 'mixed'
// intermediate. d_ws holds the f16-packed weight (512 KiB).
// v2: double-buffered B fragments so ds_load_b128 of step s+1 overlaps the
// v_wmma of step s (partial s_wait_dscnt instead of a full drain per step).

typedef __attribute__((ext_vector_type(16))) _Float16 v16h;
typedef __attribute__((ext_vector_type(8)))  _Float16 v8h;
typedef __attribute__((ext_vector_type(4)))  _Float16 v4h;
typedef __attribute__((ext_vector_type(8)))  float    v8f;
typedef __attribute__((ext_vector_type(4)))  float    v4f;

namespace {
constexpr int   kBatch  = 32;
constexpr int   kC      = 512;     // channels (M and K of the GEMM)
constexpr int   kT      = 2048;    // time steps (N of the GEMM)
constexpr int   kK      = 512;
constexpr int   kTChunk = 16;      // one WMMA N-tile per chunk
constexpr int   kWaves  = 4;       // 128 threads/block, wave32
constexpr int   kRowPad = 520;     // halfs per sB row; 520*2/16 = 65 (odd) ->
                                   // conflict-free b128 B-fragment reads
constexpr float kBeta = 0.9f;
constexpr float kTh   = 0.75f;
}

// Pack weight f32 -> f16 in the per-lane A-fragment order for
// v_wmma_f32_16x16x32_f16 (ISA 7.12.2, 16-bit A 16x32):
//   lane<16 : M=lane,    halfs j=0..7 -> K=j,    j=8..15 -> K=j+8   (0..7,16..23)
//   lane>=16: M=lane-16, halfs j=0..7 -> K=j+8,  j=8..15 -> K=j+16  (8..15,24..31)
// Linear index: (((mb*16 + s)*32 + lane)*16 + j), mb = 16-row block, s = K/32.
__global__ void snn_pack_weight_f16(const float* __restrict__ w,
                                    _Float16* __restrict__ wp) {
  int idx = blockIdx.x * blockDim.x + threadIdx.x;   // 0 .. 512*512-1
  int j   = idx & 15;
  int l   = (idx >> 4) & 31;
  int s   = (idx >> 9) & 15;
  int mb  = idx >> 13;
  int row = mb * 16 + (l & 15);
  int kl  = (j < 8 ? j : j + 8) + ((l >> 4) << 3);
  int k   = s * 32 + kl;
  wp[idx] = (_Float16)w[row * kK + k];
}

__global__ __launch_bounds__(128) void snn_fused_kernel(
    const float* __restrict__ spikes,     // [32,512,2048]
    const _Float16* __restrict__ wpk,     // packed A fragments
    const float* __restrict__ bias,       // [512]
    float* __restrict__ out)              // [32,512,2048]
{
  const int tid  = threadIdx.x;
  const int wave = tid >> 5;
  const int lane = tid & 31;
  const int cgrp = blockIdx.x;                // 0..7  (channel group of 64)
  const int b    = blockIdx.y;                // 0..31 (batch)
  const int mb   = cgrp * kWaves + wave;      // 16-channel tile id, 0..31

  __shared__ _Float16 sB[kTChunk * kRowPad];  // transposed spike slab [t][k]
  __shared__ float    sMix[kWaves][16][17];   // per-wave C tile staging

  // Preload this wave's A fragments once: 16 x v16h = 128 VGPRs/lane,
  // reused across all 128 T-chunks (W rows stay register resident).
  v16h wA[16];
  {
    const v16h* wp = (const v16h*)wpk + (size_t)(mb * 16) * 32 + lane;
#pragma unroll
    for (int s = 0; s < 16; ++s) wA[s] = wp[s * 32];
  }

  // LIF scan state: lanes 0-15 of each wave each own one channel chain.
  const int   m    = lane & 15;
  const int   ch   = mb * 16 + m;
  const float bia  = bias[ch];
  float       memv = 0.0f;
  float* outp = out + ((size_t)b * kC + ch) * (size_t)kT;
  const float* spb = spikes + (size_t)b * kC * kT;

  const int tt = tid & 15;           // t within chunk (coalesced global reads)
  const int kb = (tid >> 4) << 2;    // 4 consecutive k rows per thread

  for (int t0 = 0; t0 < kT; t0 += kTChunk) {
    __syncthreads();  // previous chunk's WMMA reads of sB are complete

    // Stage spikes[b, :, t0..t0+15] transposed into LDS as f16.
    // Reads: lanes 0-15 cover 64B contiguous per k row. Stores: v4h (b64),
    // banks 4t..4t+3 -> conflict-free across the wave.
    for (int k = kb; k < kK; k += 32) {
      float s0 = spb[(size_t)(k + 0) * kT + t0 + tt];
      float s1 = spb[(size_t)(k + 1) * kT + t0 + tt];
      float s2 = spb[(size_t)(k + 2) * kT + t0 + tt];
      float s3 = spb[(size_t)(k + 3) * kT + t0 + tt];
      __builtin_prefetch(&spb[(size_t)k * kT + t0 + tt + kTChunk], 0, 1);
      v4h pk = {(_Float16)s0, (_Float16)s1, (_Float16)s2, (_Float16)s3};
      *(v4h*)&sB[tt * kRowPad + k] = pk;
    }
    __syncthreads();

    // GEMM tile: C(16x16) += A(16x512) x B(512x16), 16 WMMA steps of K=32.
    // B fragment (ISA 7.12.2): lane column N = lane%16, K ranges
    // (lane/16)*8 .. +7 and 16+(lane/16)*8 .. +7 -> two aligned v8h LDS reads.
    // Double-buffered: B loads for step s+1 are issued before the WMMA of
    // step s, so the compiler can use partial dscnt waits and overlap the
    // LDS reads with the matrix pipe.
    v8f acc = {};
    const int nCol  = lane & 15;
    const int kHalf = (lane >> 4) << 3;
    const _Float16* bp0 = &sB[nCol * kRowPad + kHalf];

    union BU { v16h v; v8h h[2]; };
    BU bu[2];
    bu[0].h[0] = *(const v8h*)bp0;
    bu[0].h[1] = *(const v8h*)(bp0 + 16);
#pragma unroll
    for (int s = 0; s < 16; ++s) {
      if (s < 15) {
        const _Float16* bp = bp0 + (s + 1) * 32;
        bu[(s + 1) & 1].h[0] = *(const v8h*)bp;
        bu[(s + 1) & 1].h[1] = *(const v8h*)(bp + 16);
      }
      acc = __builtin_amdgcn_wmma_f32_16x16x32_f16(
          /*neg_a=*/false, wA[s], /*neg_b=*/false, bu[s & 1].v,
          /*c_mod=*/(short)0, acc, /*reuse_a=*/false, /*reuse_b=*/false);
    }

    // C layout: lane l holds M = r + 8*(l/16) in VGPR r, N = l%16.
    const int rbase = (lane >> 4) << 3;
#pragma unroll
    for (int r = 0; r < 8; ++r) sMix[wave][rbase + r][nCol] = acc[r];

    // Sequential LIF over the 16 new time steps. Same-wave LDS RAW on sMix:
    // DS ops are in-order per wave + compiler-inserted s_wait_dscnt.
    if (lane < 16) {
      v4f pk = {};
#pragma unroll
      for (int n = 0; n < 16; ++n) {
        float x = sMix[wave][m][n] + bia;
        float r = (memv > kTh) ? kTh : 0.0f;   // reset from PREVIOUS mem
        memv = kBeta * memv + x - r;           // subtract-reset Leaky
        pk[n & 3] = (memv > kTh) ? 1.0f : 0.0f;
        if ((n & 3) == 3) *(v4f*)(outp + t0 + n - 3) = pk;
      }
    }
  }
}

extern "C" void kernel_launch(void* const* d_in, const int* in_sizes, int n_in,
                              void* d_out, int out_size, void* d_ws, size_t ws_size,
                              hipStream_t stream) {
  const float* spikes = (const float*)d_in[0];   // [32,512,2048]
  const float* weight = (const float*)d_in[1];   // [512,512]
  const float* bias   = (const float*)d_in[2];   // [512]
  float*       out    = (float*)d_out;
  _Float16*    wpk    = (_Float16*)d_ws;         // 512 KiB of scratch

  (void)in_sizes; (void)n_in; (void)out_size; (void)ws_size;

  // 1) Pack weight into WMMA A-fragment order (f32 -> f16), 512*512 halfs.
  snn_pack_weight_f16<<<(kC * kK) / 256, 256, 0, stream>>>(weight, wpk);

  // 2) Fused mix + LIF scan: grid = 8 channel groups x 32 batches,
  //    128 threads = 4 wave32 waves, one 16-channel WMMA tile per wave.
  dim3 grid(8, kBatch);
  snn_fused_kernel<<<grid, 128, 0, stream>>>(spikes, wpk, bias, out);
}
